// GraphSAGEModel_64759516889909
// MI455X (gfx1250) — compile-verified, matrix operand order checked by reading, not run on hardware
//
#include <hip/hip_runtime.h>

#define NODES 50000
#define FEAT  128

typedef __attribute__((ext_vector_type(16))) _Float16 v16h;
typedef __attribute__((ext_vector_type(8)))  _Float16 v8h;
typedef __attribute__((ext_vector_type(8)))  float    v8f;

// ---------------------------------------------------------------- utilities
__global__ void zero_f32_kernel(float* __restrict__ p, int n) {
    int i = blockIdx.x * blockDim.x + threadIdx.x;
    if (i < n) p[i] = 0.0f;
}

// ------------------------------------------------------------- degree count
__global__ void degree_kernel(const int* __restrict__ dst,
                              float* __restrict__ cnt, int E) {
    int e = blockIdx.x * blockDim.x + threadIdx.x;
    if (e < E) {
        __hip_atomic_fetch_add(&cnt[dst[e]], 1.0f,
                               __ATOMIC_RELAXED, __HIP_MEMORY_SCOPE_AGENT);
    }
}

// -------------------------------------------------- scatter-add (1 wave/edge)
// lane l handles features [4l, 4l+3]; 32 lanes * 4 = 128 features.
// Hot loop of the whole model (~0.8 GB RMW traffic per layer) -> force the
// native L2-side f32 atomic (no-return form, STOREcnt-tracked).
__device__ __forceinline__ void atomic_add_f32_native(float* p, float v) {
    asm volatile("global_atomic_add_f32 %0, %1, off"
                 :
                 : "v"(p), "v"(v)
                 : "memory");
}

__global__ __launch_bounds__(256)
void scatter_add_kernel(const float* __restrict__ h,
                        const int* __restrict__ src,
                        const int* __restrict__ dst,
                        float* __restrict__ agg, int E) {
    int gid  = blockIdx.x * blockDim.x + threadIdx.x;
    int e    = gid >> 5;
    int lane = gid & 31;
    if (e >= E) return;
    int s = src[e];
    int d = dst[e];
    const float4* hs = (const float4*)(h + (size_t)s * FEAT);
    float4 v = hs[lane];
    float* ap = agg + (size_t)d * FEAT + lane * 4;
    atomic_add_f32_native(ap + 0, v.x);
    atomic_add_f32_native(ap + 1, v.y);
    atomic_add_f32_native(ap + 2, v.z);
    atomic_add_f32_native(ap + 3, v.w);
}

// ------------------------------------------ pack [mean || h_prev] -> f16 K=256
__global__ void pack_hcat_kernel(const float* __restrict__ agg,
                                 const float* __restrict__ cnt,
                                 const float* __restrict__ hprev,
                                 _Float16* __restrict__ hcat, int n /* N*FEAT */) {
    int i = blockIdx.x * blockDim.x + threadIdx.x;
    if (i >= n) return;
    int row = i >> 7;
    int k   = i & 127;
    float c = cnt[row];
    c = c > 1.0f ? c : 1.0f;
    hcat[(size_t)row * 256 + k]       = (_Float16)(agg[i] / c);
    hcat[(size_t)row * 256 + 128 + k] = (_Float16)hprev[i];
}

// ------------------------------------------------------- weight packers (f16)
__global__ void pack_w2_kernel(const float* __restrict__ Wl,
                               const float* __restrict__ Wr,
                               _Float16* __restrict__ Wp, int n /* 128*256 */) {
    int i = blockIdx.x * blockDim.x + threadIdx.x;
    if (i >= n) return;
    int row = i >> 8;     // output channel
    int k   = i & 255;    // fused K
    float v = (k < 128) ? Wl[row * 128 + k] : Wr[row * 128 + (k - 128)];
    Wp[i] = (_Float16)v;
}

__global__ void pack_f16_kernel(const float* __restrict__ src,
                                _Float16* __restrict__ dstp, int n) {
    int i = blockIdx.x * blockDim.x + threadIdx.x;
    if (i < n) dstp[i] = (_Float16)src[i];
}

// -------------------------------------------------------------- WMMA GEMM
// out[m][n] = act( sum_k A[m][k] * W[n][k] + bias[n] )
// Block = 64 rows x 128 cols. The 64xK A-tile is staged once into LDS
// (contiguous flat copy; 32 KB @ K=256). Each of the 8 waves owns 16 output
// columns and keeps 4 independent 16x16 accumulators (4 M-tiles).
// K is a compile-time constant (256 or 128): the MAC phase is fully unrolled
// straight-line code (K/32 * 4 WMMAs), all ds/global addresses fold into
// immediate offset fields, and the scheduler can hoist next-step loads across
// the current step's WMMAs. EXEC is all-ones through every WMMA; tail guards
// are epilogue-only (50000 = 781*64 + 16).
template <int K, bool RELU>
__global__ __launch_bounds__(256)
void sage_gemm_wmma_kernel(const _Float16* __restrict__ A,   // [rows][K]
                           const _Float16* __restrict__ W,   // [128][K]
                           const float* __restrict__ bias,   // [128]
                           float* __restrict__ out,          // [rows][128]
                           int rows) {
    __shared__ __align__(16) _Float16 Atile[64 * 256];

    const int tid = threadIdx.x;
    const int m0  = blockIdx.x * 64;
    const int rowsIn = (rows - m0) < 64 ? (rows - m0) : 64;

    // cooperative flat copy of the A tile (rows are contiguous: stride == K)
    {
        const uint4* g = (const uint4*)(A + (size_t)m0 * K);
        uint4*       l = (uint4*)Atile;
        const int nchunks = rowsIn * K / 8;   // uint4 == 8 halves
        for (int i = tid; i < nchunks; i += 256) l[i] = g[i];
    }
    __syncthreads();

    const int lane = tid & 31;
    const int wave = tid >> 5;     // 0..7 -> 16-column slice
    const int r    = lane & 15;    // row (A) / col (B) within tile
    const int gsel = lane >> 4;    // K half-group select
    const int n0   = wave * 16;

    const _Float16* wrow = W + (size_t)(n0 + r) * K + gsel * 8;

    v8f acc[4] = {};
    #pragma unroll
    for (int kb = 0; kb < K; kb += 32) {
        // B fragment: 2 x 16B global loads (const offsets), reused 4x
        v8h blo = *(const v8h*)(wrow + kb);
        v8h bhi = *(const v8h*)(wrow + kb + 16);
        v16h b;
        #pragma unroll
        for (int e = 0; e < 8; ++e) { b[e] = blo[e]; b[e + 8] = bhi[e]; }

        #pragma unroll
        for (int t = 0; t < 4; ++t) {
            const _Float16* arow = &Atile[(t * 16 + r) * K + gsel * 8];
            v8h alo = *(const v8h*)(arow + kb);        // ds_load_b128 offset:kb*2
            v8h ahi = *(const v8h*)(arow + kb + 16);   // ds_load_b128 offset:kb*2+32
            v16h a;
            #pragma unroll
            for (int e = 0; e < 8; ++e) { a[e] = alo[e]; a[e + 8] = ahi[e]; }
            acc[t] = __builtin_amdgcn_wmma_f32_16x16x32_f16(
                         /*neg_a=*/false, a, /*neg_b=*/false, b,
                         /*c_mod=*/(short)0, acc[t],
                         /*reuse_a=*/false, /*reuse_b=*/false);
        }
    }

    // C/D layout: lane (L&15) = N col; VGPR v holds M = v + 8*(L>>4).
    const int   ncol = n0 + r;
    const float bv   = bias[ncol];
    #pragma unroll
    for (int t = 0; t < 4; ++t) {
        #pragma unroll
        for (int v = 0; v < 8; ++v) {
            int mrow = m0 + t * 16 + v + 8 * gsel;
            if (mrow < rows) {
                float val = acc[t][v] + bv;
                if (RELU) val = fmaxf(val, 0.0f);
                out[(size_t)mrow * FEAT + ncol] = val;
            }
        }
    }
}

// ---------------------------------------------------------------- launcher
extern "C" void kernel_launch(void* const* d_in, const int* in_sizes, int n_in,
                              void* d_out, int out_size, void* d_ws, size_t ws_size,
                              hipStream_t stream) {
    const float* x   = (const float*)d_in[0];
    const int*   ei  = (const int*)d_in[1];
    const int    E   = in_sizes[1] / 2;
    const int*   src = ei;
    const int*   dst = ei + E;

    const float* Wl[4] = {(const float*)d_in[2],  (const float*)d_in[5],
                          (const float*)d_in[8],  (const float*)d_in[11]};
    const float* Wr[4] = {(const float*)d_in[3],  (const float*)d_in[6],
                          (const float*)d_in[9],  (const float*)d_in[12]};
    const float* bs[4] = {(const float*)d_in[4],  (const float*)d_in[7],
                          (const float*)d_in[10], (const float*)d_in[13]};
    const float* Wfc = (const float*)d_in[14];
    const float* bfc = (const float*)d_in[15];
    float*       out = (float*)d_out;

    // workspace carve-out
    char*  ws  = (char*)d_ws;
    size_t off = 0;
    auto carve = [&](size_t bytes) -> void* {
        void* p = ws + off;
        off += (bytes + 255) & ~(size_t)255;
        return p;
    };
    float*    agg  = (float*)   carve((size_t)NODES * FEAT * sizeof(float));
    float*    cnt  = (float*)   carve((size_t)NODES * sizeof(float));
    float*    hA   = (float*)   carve((size_t)NODES * FEAT * sizeof(float));
    float*    hB   = (float*)   carve((size_t)NODES * FEAT * sizeof(float));
    _Float16* hcat = (_Float16*)carve((size_t)NODES * 2 * FEAT * sizeof(_Float16));
    _Float16* Wp   = (_Float16*)carve((size_t)2 * FEAT * FEAT * sizeof(_Float16));

    const int NF = NODES * FEAT;              // 6,400,000
    const int gemm_blocks = (NODES + 63) / 64;
    dim3 b256(256);

    // degrees (graph fixed across layers)
    zero_f32_kernel<<<(NODES + 255) / 256, b256, 0, stream>>>(cnt, NODES);
    degree_kernel<<<(E + 255) / 256, b256, 0, stream>>>(dst, cnt, E);

    const float* hcur    = x;
    float*       hout[4] = {hA, hB, hA, hB};

    for (int L = 0; L < 4; ++L) {
        zero_f32_kernel<<<(NF + 255) / 256, b256, 0, stream>>>(agg, NF);
        scatter_add_kernel<<<(E * 32 + 255) / 256, b256, 0, stream>>>(
            hcur, src, dst, agg, E);
        pack_hcat_kernel<<<(NF + 255) / 256, b256, 0, stream>>>(
            agg, cnt, hcur, hcat, NF);
        pack_w2_kernel<<<(2 * FEAT * FEAT + 255) / 256, b256, 0, stream>>>(
            Wl[L], Wr[L], Wp, 2 * FEAT * FEAT);
        sage_gemm_wmma_kernel<256, true><<<gemm_blocks, b256, 0, stream>>>(
            hcat, Wp, bs[L], hout[L], NODES);
        hcur = hout[L];
    }

    // final fc: out = h @ Wfc.T + bfc (no relu), K = 128
    pack_f16_kernel<<<(NF + 255) / 256, b256, 0, stream>>>(hcur, hcat, NF);
    pack_f16_kernel<<<(FEAT * FEAT + 255) / 256, b256, 0, stream>>>(
        Wfc, Wp, FEAT * FEAT);
    sage_gemm_wmma_kernel<128, false><<<gemm_blocks, b256, 0, stream>>>(
        hcat, Wp, bfc, out, NODES);
}
